// MultiHeadAttention_66709432041594
// MI455X (gfx1250) — matrix operand order repaired and checked by reading, not compiled, not run on hardware
//
#include <hip/hip_runtime.h>

// ---------------------------------------------------------------------------
// MHA forward for MI455X (gfx1250, wave32, WMMA).
// bf16 V_WMMA_F32_16X16X32_BF16 for all GEMMs + flash attention.
// Async global->LDS staging (ASYNCcnt) with double-buffered tiles.
// ---------------------------------------------------------------------------

typedef __bf16 bf16x16 __attribute__((ext_vector_type(16)));
typedef __bf16 bf16x8  __attribute__((ext_vector_type(8)));
typedef float  f32x8   __attribute__((ext_vector_type(8)));

#define B_  4
#define S_  2048
#define D_  1024
#define NH_ 16
#define HD_ 64

// ---------------- async global->LDS copy (16B per lane) --------------------
#if __has_builtin(__builtin_amdgcn_global_load_async_to_lds_b128)
#define HAVE_ASYNC 1
typedef int v4i_ __attribute__((__vector_size__(4 * sizeof(int))));
typedef __attribute__((address_space(1))) v4i_* gvec_p;   // global
typedef __attribute__((address_space(3))) v4i_* svec_p;   // LDS
__device__ __forceinline__ void async_cp16(void* lds, const void* g) {
  __builtin_amdgcn_global_load_async_to_lds_b128((gvec_p)(void*)g, (svec_p)lds, 0, 0);
}
__device__ __forceinline__ void async_wait() {
#if __has_builtin(__builtin_amdgcn_s_wait_asynccnt)
  __builtin_amdgcn_s_wait_asynccnt(0);
#else
  asm volatile("s_wait_asynccnt 0x0" ::: "memory");
#endif
}
#else
#define HAVE_ASYNC 0
__device__ __forceinline__ void async_cp16(void* lds, const void* g) {
  *(uint4*)lds = *(const uint4*)g;   // sync fallback: global->VGPR->LDS
}
__device__ __forceinline__ void async_wait() {}
#endif

union FragBF { bf16x16 v; bf16x8 h[2]; };

// A-operand fragment (16x32, M x K), CDNA5 ISA 7.12.2:
// lanes 0-15: row m=lane, k={0..7,16..23}; lanes 16-31: row m=lane-16, k={8..15,24..31}
__device__ __forceinline__ bf16x16 ld_fragA(const __bf16* base, int ld) {
  int lane = threadIdx.x & 31;
  const __bf16* p = base + (lane & 15) * ld + ((lane >> 4) << 3);
  FragBF f;
  f.h[0] = *(const bf16x8*)(p);
  f.h[1] = *(const bf16x8*)(p + 16);
  return f.v;
}

// B-operand fragment (32x16, K x N): lanes 0-15: col n=lane, k={0..7,16..23};
// lanes 16-31: col n=lane-16, k={8..15,24..31}. Read from B^T storage (n-major, k contig).
__device__ __forceinline__ bf16x16 ld_fragB(const __bf16* base, int ld) {
  int lane = threadIdx.x & 31;
  const __bf16* p = base + (lane & 15) * ld + ((lane >> 4) << 3);
  FragBF f;
  f.h[0] = *(const bf16x8*)(p);
  f.h[1] = *(const bf16x8*)(p + 16);
  return f.v;
}

__device__ __forceinline__ f32x8 wmma_bf16(bf16x16 a, bf16x16 b, f32x8 c) {
  return __builtin_amdgcn_wmma_f32_16x16x32_bf16(false, a, false, b, (short)0, c,
                                                 false, false);
}

// ---------------------------------------------------------------------------
__global__ __launch_bounds__(256)
void cvt_f32_bf16(const float* __restrict__ in, __bf16* __restrict__ out, int n) {
  int i = blockIdx.x * 256 + threadIdx.x;
  int stride = gridDim.x * 256;
  for (; i < n; i += stride) out[i] = (__bf16)in[i];
}

// ---------------------------------------------------------------------------
// C[m,n] = sum_k A[m,k] * W[n,k]   (A: MxK bf16 row-major, W: NxK bf16 row-major)
// MODE 0: bf16 scattered to [B, NH, S, HD] (Q/K/V).  MODE 1: f32 [M, N].
template <int MODE>
__global__ __launch_bounds__(256)
void gemm_xWT(const __bf16* __restrict__ A, const __bf16* __restrict__ W,
              void* __restrict__ outp) {
  __shared__ __bf16 sA[2][128 * 32];
  __shared__ __bf16 sB[2][128 * 32];
  const int K = D_;
  const int bm = blockIdx.y * 128;
  const int bn = blockIdx.x * 128;
  const int tid = threadIdx.x;
  const int wid = tid >> 5;
  const int wm = wid >> 1;   // 0..3 : 32 rows
  const int wn = wid & 1;    // 0..1 : 64 cols

  auto stage = [&](int buf, int kt) {
#pragma unroll
    for (int i = 0; i < 2; ++i) {  // 512 x 16B chunks per tile, 2 per thread
      int f = tid + i * 256;
      int row = f >> 2;
      int kp = f & 3;
      async_cp16(&sA[buf][row * 32 + kp * 8],
                 A + (size_t)(bm + row) * K + kt + kp * 8);
      async_cp16(&sB[buf][row * 32 + kp * 8],
                 W + (size_t)(bn + row) * K + kt + kp * 8);
    }
  };

  f32x8 acc[2][4];
  const f32x8 zero = {0.f, 0.f, 0.f, 0.f, 0.f, 0.f, 0.f, 0.f};
#pragma unroll
  for (int i = 0; i < 2; ++i)
#pragma unroll
    for (int j = 0; j < 4; ++j) acc[i][j] = zero;

  stage(0, 0);
  int cur = 0;
  for (int kt = 0; kt < K; kt += 32) {
    async_wait();
    __syncthreads();
    if (kt + 32 < K) stage(cur ^ 1, kt + 32);  // overlap DMA with WMMA

    bf16x16 af[2], bfm[4];
#pragma unroll
    for (int mt = 0; mt < 2; ++mt)
      af[mt] = ld_fragA(&sA[cur][(wm * 32 + mt * 16) * 32], 32);
#pragma unroll
    for (int nt = 0; nt < 4; ++nt)
      bfm[nt] = ld_fragB(&sB[cur][(wn * 64 + nt * 16) * 32], 32);
#pragma unroll
    for (int mt = 0; mt < 2; ++mt)
#pragma unroll
      for (int nt = 0; nt < 4; ++nt)
        acc[mt][nt] = wmma_bf16(af[mt], bfm[nt], acc[mt][nt]);
    cur ^= 1;
  }

  // C fragment: lane l, elem j -> row = j + 8*(l>>4), col = l&15
  const int lane = tid & 31;
  const int col0 = lane & 15;
  const int rhi = (lane >> 4) * 8;
#pragma unroll
  for (int mt = 0; mt < 2; ++mt)
#pragma unroll
    for (int nt = 0; nt < 4; ++nt)
#pragma unroll
      for (int j = 0; j < 8; ++j) {
        int m = bm + wm * 32 + mt * 16 + j + rhi;
        int n = bn + wn * 64 + nt * 16 + col0;
        float v = acc[mt][nt][j];
        if (MODE == 0) {
          int b = m >> 11, s = m & (S_ - 1);
          int h = n >> 6, d = n & (HD_ - 1);
          ((__bf16*)outp)[(((size_t)b * NH_ + h) * S_ + s) * HD_ + d] = (__bf16)v;
        } else {
          ((float*)outp)[(size_t)m * D_ + n] = v;
        }
      }
}

// ---------------------------------------------------------------------------
// Flash attention per (b,h). Block = 128 query rows, 8 waves x 16-row strips.
// Q/K/V bf16 [B*NH, S, 64]. Output Y bf16 [B, S, D] (col offset h*64).
__global__ __launch_bounds__(256)
void flash_attn(const __bf16* __restrict__ Q, const __bf16* __restrict__ Kc,
                const __bf16* __restrict__ V, __bf16* __restrict__ Y) {
  __shared__ __bf16 sQ[128 * 64];            // 16KB
  __shared__ __bf16 sK[2][128 * 64];         // 32KB (k rows, d contiguous)
  __shared__ __bf16 sVt[2][64 * 128];        // 32KB (V^T: d rows, k contiguous)
  __shared__ __bf16 sP[8 * 16 * 128];        // 32KB (per-wave P strips)

  const int qb = blockIdx.x;                 // 0..15
  const int bh = blockIdx.y;                 // 0..63
  const int b = bh >> 4, h = bh & 15;
  const __bf16* Qp = Q + ((size_t)bh * S_ + (size_t)qb * 128) * HD_;
  const __bf16* Kp = Kc + (size_t)bh * S_ * HD_;
  const __bf16* Vp = V + (size_t)bh * S_ * HD_;
  const int tid = threadIdx.x, wid = tid >> 5, lane = tid & 31;
  const int col0 = lane & 15, rhi = (lane >> 4) * 8;

  auto stageKV = [&](int buf, int kb) {
#pragma unroll
    for (int i = 0; i < 4; ++i) {  // K tile: 1024 x 16B chunks
      int f = tid + i * 256;
      async_cp16(&sK[buf][f * 8], Kp + (size_t)kb * 128 * HD_ + f * 8);
    }
    // V tile staged transposed (needs VGPR round trip)
    int r = tid >> 1, hf = tid & 1;
    union { uint4 q[4]; __bf16 e[32]; } tmp;
    const uint4* gp = (const uint4*)(Vp + (size_t)(kb * 128 + r) * HD_ + hf * 32);
    tmp.q[0] = gp[0]; tmp.q[1] = gp[1]; tmp.q[2] = gp[2]; tmp.q[3] = gp[3];
#pragma unroll
    for (int i = 0; i < 32; ++i) sVt[buf][(hf * 32 + i) * 128 + r] = tmp.e[i];
  };

  // stage Q tile (128x64) + first K/V tiles
#pragma unroll
  for (int i = 0; i < 4; ++i) {
    int f = tid + i * 256;
    async_cp16(&sQ[f * 8], Qp + f * 8);
  }
  stageKV(0, 0);

  const f32x8 zero = {0.f, 0.f, 0.f, 0.f, 0.f, 0.f, 0.f, 0.f};
  f32x8 accO[4];
#pragma unroll
  for (int t = 0; t < 4; ++t) accO[t] = zero;
  float m_[8], l_[8];
#pragma unroll
  for (int j = 0; j < 8; ++j) { m_[j] = -__builtin_inff(); l_[j] = 0.f; }
  const float c = 0.125f * 1.44269504088896340736f;  // (1/sqrt(64)) * log2(e)

  bf16x16 qf[2];
  int cur = 0;
  for (int kb = 0; kb < S_ / 128; ++kb) {
    async_wait();
    __syncthreads();
    if (kb == 0) {  // uniform branch; sQ ready after first barrier
      qf[0] = ld_fragA(&sQ[(wid * 16) * 64 + 0], 64);
      qf[1] = ld_fragA(&sQ[(wid * 16) * 64 + 32], 64);
    }
    if (kb + 1 < S_ / 128) stageKV(cur ^ 1, kb + 1);  // overlap with compute

    // S = Q K^T : wave computes its 16x128 strip (8 col tiles)
    f32x8 sacc[8];
#pragma unroll
    for (int t = 0; t < 8; ++t) sacc[t] = zero;
#pragma unroll
    for (int t = 0; t < 8; ++t)
#pragma unroll
      for (int kk = 0; kk < 2; ++kk) {
        bf16x16 kf = ld_fragB(&sK[cur][(t * 16) * 64 + kk * 32], 64);
        sacc[t] = wmma_bf16(qf[kk], kf, sacc[t]);
      }

    // online softmax; row r lives in elem j of lanes sharing l>>4 half
#pragma unroll
    for (int j = 0; j < 8; ++j) {
      float mx = sacc[0][j];
#pragma unroll
      for (int t = 1; t < 8; ++t) mx = fmaxf(mx, sacc[t][j]);
      for (int off = 1; off < 16; off <<= 1) mx = fmaxf(mx, __shfl_xor(mx, off, 32));
      mx *= c;
      float mn = fmaxf(m_[j], mx);
      float r = __builtin_exp2f(m_[j] - mn);
      m_[j] = mn;
      float rs = 0.f;
#pragma unroll
      for (int t = 0; t < 8; ++t) {
        float p = __builtin_exp2f(sacc[t][j] * c - mn);
        sacc[t][j] = p;
        rs += p;
      }
      for (int off = 1; off < 16; off <<= 1) rs += __shfl_xor(rs, off, 32);
      l_[j] = l_[j] * r + rs;
#pragma unroll
      for (int tt = 0; tt < 4; ++tt) accO[tt][j] *= r;
    }

    // P round-trips through a private LDS strip to reach A-fragment layout;
    // same-wave DS ordering + compiler s_wait_dscnt makes this barrier-free.
    __bf16* myP = &sP[wid * 16 * 128];
#pragma unroll
    for (int t = 0; t < 8; ++t)
#pragma unroll
      for (int j = 0; j < 8; ++j)
        myP[(j + rhi) * 128 + t * 16 + col0] = (__bf16)sacc[t][j];
    bf16x16 pf[4];
#pragma unroll
    for (int kk = 0; kk < 4; ++kk) pf[kk] = ld_fragA(&myP[kk * 32], 128);

    // O += P V
#pragma unroll
    for (int tt = 0; tt < 4; ++tt)
#pragma unroll
      for (int kk = 0; kk < 4; ++kk) {
        bf16x16 vf = ld_fragB(&sVt[cur][(tt * 16) * 128 + kk * 32], 128);
        accO[tt] = wmma_bf16(pf[kk], vf, accO[tt]);
      }
    cur ^= 1;
  }

  // normalize and write Y[b, s, h*64 + d]
  const size_t ybase = ((size_t)b * S_) * D_ + (size_t)h * HD_;
#pragma unroll
  for (int j = 0; j < 8; ++j) {
    float inv = 1.f / l_[j];
    int s = qb * 128 + wid * 16 + j + rhi;
#pragma unroll
    for (int tt = 0; tt < 4; ++tt) {
      int d = tt * 16 + col0;
      Y[ybase + (size_t)s * D_ + d] = (__bf16)(accO[tt][j] * inv);
    }
  }
}

// ---------------------------------------------------------------------------
extern "C" void kernel_launch(void* const* d_in, const int* in_sizes, int n_in,
                              void* d_out, int out_size, void* d_ws, size_t ws_size,
                              hipStream_t stream) {
  const float* x  = (const float*)d_in[0];
  const float* Wq = (const float*)d_in[1];
  const float* Wk = (const float*)d_in[2];
  const float* Wv = (const float*)d_in[3];
  const float* Wo = (const float*)d_in[4];

  const size_t nx = (size_t)B_ * S_ * D_;  // 8M elems
  const size_t nw = (size_t)D_ * D_;       // 1M elems

  char* ws = (char*)d_ws;
  __bf16* xb  = (__bf16*)ws; ws += nx * 2;   // reused as Y later
  __bf16* wqb = (__bf16*)ws; ws += nw * 2;
  __bf16* wkb = (__bf16*)ws; ws += nw * 2;
  __bf16* wvb = (__bf16*)ws; ws += nw * 2;
  __bf16* wob = (__bf16*)ws; ws += nw * 2;
  __bf16* Qb  = (__bf16*)ws; ws += nx * 2;
  __bf16* Kb  = (__bf16*)ws; ws += nx * 2;
  __bf16* Vb  = (__bf16*)ws; ws += nx * 2;
  __bf16* Yb  = xb;  // x not needed once Q/K/V are projected

  cvt_f32_bf16<<<2048, 256, 0, stream>>>(x, xb, (int)nx);
  cvt_f32_bf16<<<512, 256, 0, stream>>>(Wq, wqb, (int)nw);
  cvt_f32_bf16<<<512, 256, 0, stream>>>(Wk, wkb, (int)nw);
  cvt_f32_bf16<<<512, 256, 0, stream>>>(Wv, wvb, (int)nw);
  cvt_f32_bf16<<<512, 256, 0, stream>>>(Wo, wob, (int)nw);

  dim3 gg(D_ / 128, (B_ * S_) / 128);  // (8, 64)
  gemm_xWT<0><<<gg, 256, 0, stream>>>(xb, wqb, Qb);
  gemm_xWT<0><<<gg, 256, 0, stream>>>(xb, wkb, Kb);
  gemm_xWT<0><<<gg, 256, 0, stream>>>(xb, wvb, Vb);

  flash_attn<<<dim3(S_ / 128, B_ * NH_), 256, 0, stream>>>(Qb, Kb, Vb, Yb);

  gemm_xWT<1><<<gg, 256, 0, stream>>>(Yb, wob, d_out);
}
